// GraphAggregationLayer_11287174054105
// MI455X (gfx1250) — matrix operand order repaired and checked by reading, not compiled, not run on hardware
//
#include <hip/hip_runtime.h>
#include <hip/hip_bf16.h>

typedef __attribute__((ext_vector_type(16))) _Float16 v16h;
typedef __attribute__((ext_vector_type(8)))  float    v8f;

#define WMMA_F16(A, B, C) \
    __builtin_amdgcn_wmma_f32_16x16x32_f16(false, (A), false, (B), (short)0, (C), false, false)

// ---------- order-preserving float<->uint for atomic max ----------
__device__ __forceinline__ unsigned f2ord(float f) {
    unsigned u = __float_as_uint(f);
    return (u & 0x80000000u) ? ~u : (u | 0x80000000u);
}
__device__ __forceinline__ float ord2f(unsigned u) {
    return __uint_as_float((u & 0x80000000u) ? (u & 0x7fffffffu) : ~u);
}

// ---------- WMMA fragment loaders (CDNA5 16x16x32 f16 layouts) ----------
// A-matrix 16x32: row = lane&15; lanes 0-15 hold K = k0+[0..7] and k0+16+[0..7],
// lanes 16-31 hold K = k0+8+[0..7] and k0+24+[0..7].
__device__ __forceinline__ v16h load_a_f32(const float* row, int k0, int lane) {
    int half = (lane >> 4) & 1;
    const float* p0 = row + k0 + half * 8;
    const float* p1 = row + k0 + 16 + half * 8;
    v16h a;
#pragma unroll
    for (int e = 0; e < 8; ++e) { a[e] = (_Float16)p0[e]; a[e + 8] = (_Float16)p1[e]; }
    return a;
}

__device__ __forceinline__ v16h load_a_f32_scaled(const float* row, int k0, int lane,
                                                  const float* inv_per_head) {
    int half = (lane >> 4) & 1;
    int kb0 = k0 + half * 8;
    int kb1 = k0 + 16 + half * 8;
    float s0 = inv_per_head[kb0 >> 4];
    float s1 = inv_per_head[kb1 >> 4];
    v16h a;
#pragma unroll
    for (int e = 0; e < 8; ++e) {
        a[e]     = (_Float16)(row[kb0 + e] * s0);
        a[e + 8] = (_Float16)(row[kb1 + e] * s1);
    }
    return a;
}

__device__ __forceinline__ v16h load_a_f16(const _Float16* row, int k0, int lane) {
    int half = (lane >> 4) & 1;
    const _Float16* p0 = row + k0 + half * 8;
    const _Float16* p1 = row + k0 + 16 + half * 8;
    v16h a;
#pragma unroll
    for (int e = 0; e < 8; ++e) { a[e] = p0[e]; a[e + 8] = p1[e]; }
    return a;
}

// B-matrix 32x16 from transposed f16 weight Wt[dout][din]:
// col = lane&15, lanes 0-15 hold K = k0+[0..7] and k0+16+[0..7],
// lanes 16-31 hold K = k0+8+[0..7] and k0+24+[0..7].
__device__ __forceinline__ v16h load_b_wt(const _Float16* Wt, int din, int ncol, int k0, int lane) {
    const _Float16* p = Wt + (size_t)ncol * din + k0 + ((lane >> 4) & 1) * 8;
    v16h b;
#pragma unroll
    for (int e = 0; e < 8; ++e) { b[e] = p[e]; b[e + 8] = p[16 + e]; }
    return b;
}

// =====================================================================
// Kernel 0: transpose+convert weights to f16, init softmax stats
// =====================================================================
__global__ __launch_bounds__(256) void convert_weights(
    const float* Wq, const float* Wk, const float* Wv, const float* Wo,
    const float* W1, const float* W2,
    _Float16* wqT, _Float16* wkT, _Float16* wvT, _Float16* woT,
    _Float16* w1T, _Float16* w2T,
    unsigned* gmaxbits, float* gsum) {
    int id = blockIdx.x * 256 + threadIdx.x;
    if (blockIdx.x == 0 && threadIdx.x < 4) {
        gmaxbits[threadIdx.x] = f2ord(-3.402823466e38f);
        gsum[threadIdx.x] = 0.0f;
    }
    if (id < 4 * 4096) {                       // four 64x64 matrices
        int w = id >> 12, o = id & 4095;
        int n = o >> 6, kk = o & 63;
        const float* W = (w == 0) ? Wq : (w == 1) ? Wk : (w == 2) ? Wv : Wo;
        _Float16* T   = (w == 0) ? wqT : (w == 1) ? wkT : (w == 2) ? wvT : woT;
        T[o] = (_Float16)W[kk * 64 + n];
    } else if (id < 4 * 4096 + 16384) {        // W1: 64x256 -> w1T[256][64]
        int o = id - 4 * 4096;
        int n = o >> 6, kk = o & 63;
        w1T[o] = (_Float16)W1[kk * 256 + n];
    } else if (id < 4 * 4096 + 32768) {        // W2: 256x64 -> w2T[64][256]
        int o = id - 4 * 4096 - 16384;
        int n = o >> 8, kk = o & 255;
        w2T[o] = (_Float16)W2[kk * 64 + n];
    }
}

// =====================================================================
// Kernel 1: QKV projections via WMMA. One 16-row tile per wave.
// =====================================================================
__global__ __launch_bounds__(64) void qkv_kernel(
    const float* __restrict__ x,
    const _Float16* __restrict__ wqT, const float* __restrict__ bq,
    const _Float16* __restrict__ wkT, const float* __restrict__ bk,
    const _Float16* __restrict__ wvT, const float* __restrict__ bv,
    float* __restrict__ qb, float* __restrict__ kb, float* __restrict__ vb,
    int Nn) {
    int lane = threadIdx.x & 31;
    int wave = threadIdx.x >> 5;
    int tile = blockIdx.x * 2 + wave;
    int m0 = tile * 16;
    if (m0 >= Nn) return;                       // wave-uniform
    int nl = lane & 15;
    int half = (lane >> 4) & 1;

    const float* arow = x + (size_t)(m0 + nl) * 64;
    v16h a0 = load_a_f32(arow, 0, lane);
    v16h a1 = load_a_f32(arow, 32, lane);

    const _Float16* Wts[3] = {wqT, wkT, wvT};
    const float*    bs[3]  = {bq, bk, bv};
    float*          outs[3] = {qb, kb, vb};
#pragma unroll
    for (int m = 0; m < 3; ++m) {
#pragma unroll
        for (int nt = 0; nt < 4; ++nt) {
            int col = nt * 16 + nl;
            v8f c = {};
            c = WMMA_F16(a0, load_b_wt(Wts[m], 64, col, 0, lane), c);
            c = WMMA_F16(a1, load_b_wt(Wts[m], 64, col, 32, lane), c);
            float bias = bs[m][col];
            int rbase = m0 + half * 8;
#pragma unroll
            for (int r = 0; r < 8; ++r)
                outs[m][(size_t)(rbase + r) * 64 + col] = c[r] + bias;
        }
    }
}

// =====================================================================
// Kernel 2: per-edge attention scores + global per-head max
// =====================================================================
__global__ __launch_bounds__(256) void edge_scores(
    const int* __restrict__ src, const int* __restrict__ dst,
    const float* __restrict__ qb, const float* __restrict__ kb,
    float* __restrict__ scores, unsigned* gmaxbits, long long E) {
    long long e = (long long)blockIdx.x * 256 + threadIdx.x;
    float sc[4];
    bool inb = e < E;
    if (inb) {
        int s = src[e], d = dst[e];
        const float4* qp = (const float4*)(qb + (size_t)s * 64);
        const float4* kp = (const float4*)(kb + (size_t)d * 64);
#pragma unroll
        for (int h = 0; h < 4; ++h) {
            float acc = 0.0f;
#pragma unroll
            for (int j = 0; j < 4; ++j) {
                float4 a = qp[h * 4 + j];
                float4 b = kp[h * 4 + j];
                acc += a.x * b.x + a.y * b.y + a.z * b.z + a.w * b.w;
            }
            sc[h] = acc;
        }
        *(float4*)(scores + e * 4) = make_float4(sc[0], sc[1], sc[2], sc[3]);
    } else {
#pragma unroll
        for (int h = 0; h < 4; ++h) sc[h] = -3.402823466e38f;
    }
    // wave reduce max per head
#pragma unroll
    for (int off = 16; off > 0; off >>= 1) {
#pragma unroll
        for (int h = 0; h < 4; ++h)
            sc[h] = fmaxf(sc[h], __shfl_xor(sc[h], off, 32));
    }
    if ((threadIdx.x & 31) == 0) {
#pragma unroll
        for (int h = 0; h < 4; ++h) atomicMax(&gmaxbits[h], f2ord(sc[h]));
    }
}

// =====================================================================
// Kernel 3: w = exp(s - max); gsum += w; agg[dst] += w * v[src]
// =====================================================================
__global__ __launch_bounds__(256) void edge_exp_scatter(
    const int* __restrict__ src, const int* __restrict__ dst,
    const float* __restrict__ scores, const unsigned* __restrict__ gmaxbits,
    const float* __restrict__ vb, float* __restrict__ agg,
    float* __restrict__ gsum, long long E) {
    long long e = (long long)blockIdx.x * 256 + threadIdx.x;
    float mx[4];
#pragma unroll
    for (int h = 0; h < 4; ++h) mx[h] = ord2f(gmaxbits[h]);
    float w[4] = {0.f, 0.f, 0.f, 0.f};
    bool inb = e < E;
    if (inb) {
        float4 sc = *(const float4*)(scores + e * 4);
        w[0] = __expf(sc.x - mx[0]);
        w[1] = __expf(sc.y - mx[1]);
        w[2] = __expf(sc.z - mx[2]);
        w[3] = __expf(sc.w - mx[3]);
    }
    float ws[4] = {w[0], w[1], w[2], w[3]};
#pragma unroll
    for (int off = 16; off > 0; off >>= 1) {
#pragma unroll
        for (int h = 0; h < 4; ++h) ws[h] += __shfl_xor(ws[h], off, 32);
    }
    if ((threadIdx.x & 31) == 0) {
#pragma unroll
        for (int h = 0; h < 4; ++h) atomicAdd(&gsum[h], ws[h]);
    }
    if (inb) {
        int s = src[e], d = dst[e];
        const float4* vp = (const float4*)(vb + (size_t)s * 64);
        float* ap = agg + (size_t)d * 64;
#pragma unroll
        for (int h = 0; h < 4; ++h) {
#pragma unroll
            for (int j = 0; j < 4; ++j) {
                float4 vv = vp[h * 4 + j];
                int base = h * 16 + j * 4;
                atomicAdd(ap + base + 0, w[h] * vv.x);
                atomicAdd(ap + base + 1, w[h] * vv.y);
                atomicAdd(ap + base + 2, w[h] * vv.z);
                atomicAdd(ap + base + 3, w[h] * vv.w);
            }
        }
    }
}

// =====================================================================
// Kernel 4: out = (agg/sum)@Wo+bo ; ffn = gelu(out@W1+b1)@W2+b2 ;
//           final = LN(ffn + out). One 16-node tile per wave.
// =====================================================================
__global__ __launch_bounds__(64) void finalize_kernel(
    const float* __restrict__ agg, const float* __restrict__ gsum,
    const _Float16* __restrict__ woT, const float* __restrict__ bo,
    const _Float16* __restrict__ w1T, const float* __restrict__ b1,
    const _Float16* __restrict__ w2T, const float* __restrict__ b2,
    const float* __restrict__ g2, const float* __restrict__ beta2,
    float* __restrict__ out, int Nn) {
    __shared__ float    s_out[2][16][68];
    __shared__ _Float16 s_hid[2][16][264];

    int lane = threadIdx.x & 31;
    int wave = threadIdx.x >> 5;
    int tile = blockIdx.x * 2 + wave;
    int m0 = tile * 16;
    bool active = m0 < Nn;                      // wave-uniform
    int nl = lane & 15;
    int half = (lane >> 4) & 1;

    float inv[4];
#pragma unroll
    for (int h = 0; h < 4; ++h) inv[h] = 1.0f / gsum[h];

    if (active) {
        // ---- out = (agg * inv) @ Wo + bo ----
        const float* arow = agg + (size_t)(m0 + nl) * 64;
        v16h a0 = load_a_f32_scaled(arow, 0, lane, inv);
        v16h a1 = load_a_f32_scaled(arow, 32, lane, inv);
#pragma unroll
        for (int nt = 0; nt < 4; ++nt) {
            int col = nt * 16 + nl;
            v8f c = {};
            c = WMMA_F16(a0, load_b_wt(woT, 64, col, 0, lane), c);
            c = WMMA_F16(a1, load_b_wt(woT, 64, col, 32, lane), c);
            float bb = bo[col];
#pragma unroll
            for (int r = 0; r < 8; ++r) s_out[wave][half * 8 + r][col] = c[r] + bb;
        }
    }
    __syncthreads();
    if (active) {
        // ---- hid = gelu(out @ W1 + b1), 16x256, exact erf gelu ----
        const float* orow = &s_out[wave][nl][0];
        v16h h0 = load_a_f32(orow, 0, lane);
        v16h h1 = load_a_f32(orow, 32, lane);
#pragma unroll
        for (int nt = 0; nt < 16; ++nt) {
            int col = nt * 16 + nl;
            v8f c = {};
            c = WMMA_F16(h0, load_b_wt(w1T, 64, col, 0, lane), c);
            c = WMMA_F16(h1, load_b_wt(w1T, 64, col, 32, lane), c);
            float bb = b1[col];
#pragma unroll
            for (int r = 0; r < 8; ++r) {
                float xv = c[r] + bb;
                float g = 0.5f * xv * (1.0f + erff(xv * 0.70710678118f));
                s_hid[wave][half * 8 + r][col] = (_Float16)g;
            }
        }
    }
    __syncthreads();
    if (active) {
        // ---- final = hid @ W2 + b2 + out (residual) ----
        const _Float16* hrow = &s_hid[wave][nl][0];
        v16h ah[8];
#pragma unroll
        for (int ks = 0; ks < 8; ++ks) ah[ks] = load_a_f16(hrow, ks * 32, lane);
#pragma unroll
        for (int nt = 0; nt < 4; ++nt) {
            int col = nt * 16 + nl;
            v8f c = {};
#pragma unroll
            for (int ks = 0; ks < 8; ++ks)
                c = WMMA_F16(ah[ks], load_b_wt(w2T, 256, col, ks * 32, lane), c);
            float bb = b2[col];
#pragma unroll
            for (int r = 0; r < 8; ++r) {
                int rr = half * 8 + r;
                s_out[wave][rr][col] = c[r] + bb + s_out[wave][rr][col];
            }
        }
    }
    __syncthreads();
    if (active) {
        // ---- LayerNorm: 2 lanes per row, 32 cols each ----
        int row = lane >> 1;
        int hc = (lane & 1) * 32;
        float mu = 0.f, m2 = 0.f;
#pragma unroll
        for (int j = 0; j < 32; ++j) {
            float v = s_out[wave][row][hc + j];
            mu += v; m2 += v * v;
        }
        mu += __shfl_xor(mu, 1, 32);
        m2 += __shfl_xor(m2, 1, 32);
        mu *= (1.0f / 64.0f);
        float var = m2 * (1.0f / 64.0f) - mu * mu;
        float rs = rsqrtf(var + 1e-5f);
        float* orow = out + (size_t)(m0 + row) * 64 + hc;
#pragma unroll
        for (int j = 0; j < 32; ++j) {
            int col = hc + j;
            orow[j] = (s_out[wave][row][col] - mu) * rs * g2[col] + beta2[col];
        }
    }
}

// =====================================================================
extern "C" void kernel_launch(void* const* d_in, const int* in_sizes, int n_in,
                              void* d_out, int out_size, void* d_ws, size_t ws_size,
                              hipStream_t stream) {
    const float* x   = (const float*)d_in[0];
    const int*   ei  = (const int*)d_in[1];   // (2, E) int
    const float* Wq  = (const float*)d_in[2];
    const float* bq  = (const float*)d_in[3];
    const float* Wk  = (const float*)d_in[4];
    const float* bk  = (const float*)d_in[5];
    const float* Wv  = (const float*)d_in[6];
    const float* bv  = (const float*)d_in[7];
    const float* Wo  = (const float*)d_in[8];
    const float* bo  = (const float*)d_in[9];
    const float* W1  = (const float*)d_in[10];
    const float* b1  = (const float*)d_in[11];
    const float* W2  = (const float*)d_in[12];
    const float* b2  = (const float*)d_in[13];
    const float* g2  = (const float*)d_in[14];
    const float* bt2 = (const float*)d_in[15];
    float* out = (float*)d_out;

    const int Nn = in_sizes[0] / 64;
    const long long E = in_sizes[1] / 2;
    const int* src = ei;
    const int* dst = ei + E;

    // ---- workspace carve (256B aligned) ----
    char* p = (char*)d_ws;
    auto take = [&](size_t bytes) -> void* {
        void* r = (void*)p;
        p += (bytes + 255) & ~(size_t)255;
        return r;
    };
    const size_t N64 = (size_t)Nn * 64;
    float*    qb      = (float*)take(N64 * 4);
    float*    kb      = (float*)take(N64 * 4);
    float*    vb      = (float*)take(N64 * 4);
    float*    agg     = (float*)take(N64 * 4);
    float*    scores  = (float*)take((size_t)E * 4 * 4);
    _Float16* wT      = (_Float16*)take(49152 * 2);
    unsigned* gmaxbit = (unsigned*)take(64);
    float*    gsum    = (float*)take(64);

    _Float16* wqT = wT;
    _Float16* wkT = wT + 4096;
    _Float16* wvT = wT + 8192;
    _Float16* woT = wT + 12288;
    _Float16* w1T = wT + 16384;
    _Float16* w2T = wT + 32768;

    // 0) zero the aggregation buffer (must happen every call)
    hipMemsetAsync(agg, 0, N64 * 4, stream);

    // 1) weight transpose/convert + stats init
    convert_weights<<<192, 256, 0, stream>>>(Wq, Wk, Wv, Wo, W1, W2,
                                             wqT, wkT, wvT, woT, w1T, w2T,
                                             gmaxbit, gsum);

    // 2) q,k,v projections (WMMA): one 16-row tile per wave, 2 waves/block
    int tiles = (Nn + 15) / 16;
    int gblocks = (tiles + 1) / 2;
    qkv_kernel<<<gblocks, 64, 0, stream>>>(x, wqT, bq, wkT, bk, wvT, bv,
                                           qb, kb, vb, Nn);

    // 3) edge scores + global per-head max
    int eblocks = (int)((E + 255) / 256);
    edge_scores<<<eblocks, 256, 0, stream>>>(src, dst, qb, kb, scores, gmaxbit, E);

    // 4) exp weights, per-head sums, exp-weighted scatter into agg
    edge_exp_scatter<<<eblocks, 256, 0, stream>>>(src, dst, scores, gmaxbit,
                                                  vb, agg, gsum, E);

    // 5) Wo projection (with 1/sum folded in) + FFN + residual + LayerNorm
    finalize_kernel<<<gblocks, 64, 0, stream>>>(agg, gsum, woT, bo, w1T, b1,
                                                w2T, b2, g2, bt2, out, Nn);
}
